// KnnMemory_37237366456893
// MI455X (gfx1250) — matrix operand order repaired and checked by reading, not compiled, not run on hardware
//
#include <hip/hip_runtime.h>
#include <cstdint>

// ---------------- problem constants ----------------
#define NQ    256
#define DIMS  128
#define NM    500000
#define KNN   200
#define NCLS  400
#define INV_T 14.285714285714286f     // 1/0.07
#define NBINS 128                     // bins of width 1.0 over [-64, 64)
#define CAP   2048
#define MT    (NM / 16)               // 31250 tiles of 16 memory rows
#define AST   136                     // bf16 LDS row stride: 272B = 68 dwords -> bank-conflict free

static_assert(NM % 16 == 0, "memory rows must tile by 16");

typedef __attribute__((ext_vector_type(8)))  float        v8f;
typedef __attribute__((ext_vector_type(8)))  __bf16       v8bf;
typedef __attribute__((ext_vector_type(16))) __bf16       v16bf;
typedef __attribute__((ext_vector_type(4)))  int          v4i;
typedef __attribute__((ext_vector_type(4)))  unsigned int v4u;
typedef __attribute__((ext_vector_type(2)))  unsigned int v2u;

typedef __attribute__((address_space(1))) v4i* gp4;
typedef __attribute__((address_space(3))) v4i* lp4;

// ---- CDNA5 async global->LDS copy (ASYNCcnt) -- unconditional this round ----
__device__ __forceinline__ void copy16_async(const float* g, float* l) {
    __builtin_amdgcn_global_load_async_to_lds_b128((gp4)(void*)g, (lp4)l, 0, 0);
}
__device__ __forceinline__ void async_wait_le2() {
    __builtin_amdgcn_s_wait_asynccnt(2);
}
__device__ __forceinline__ void async_wait_le0() {
    __builtin_amdgcn_s_wait_asynccnt(0);
}

// ---- bf16 helpers ----
__device__ __forceinline__ unsigned short bfbits(__bf16 x) {
    unsigned short u;
    __builtin_memcpy(&u, &x, 2);
    return u;
}
__device__ __forceinline__ unsigned int pk(__bf16 a, __bf16 b) {
    return (unsigned int)bfbits(a) | ((unsigned int)bfbits(b) << 16);
}
// split fp32 -> bf16 hi + bf16 lo (residual), RNE
__device__ __forceinline__ void split2(float x, __bf16* hp, __bf16* lp) {
    __bf16 h = (__bf16)x;
    *hp = h;
    *lp = (__bf16)(x - (float)h);
}

__device__ __forceinline__ v16bf cat8(v8bf a, v8bf b) {
    v16bf r;
    #pragma unroll
    for (int i = 0; i < 8; ++i) { r[i] = a[i]; r[i + 8] = b[i]; }
    return r;
}
// A fragment (16x32): chunks at kc + 8*(lane>>4) and +16
__device__ __forceinline__ v16bf ldA(const __bf16* rowp, int kc, int lh8) {
    v8bf x = *(const v8bf*)(rowp + kc + lh8);
    v8bf y = *(const v8bf*)(rowp + kc + lh8 + 16);
    return cat8(x, y);
}
// B fragment (32x16): chunks at kc + 8*(lane>>4) and +16
__device__ __forceinline__ v16bf ldB(const __bf16* rowp, int kc, int lh16) {
    v8bf x = *(const v8bf*)(rowp + kc + (lh16 >> 1));
    v8bf y = *(const v8bf*)(rowp + kc + (lh16 >> 1) + 16);
    return cat8(x, y);
}

#define WMMA_BF16(A, B, C) \
    __builtin_amdgcn_wmma_f32_16x16x32_bf16(false, (A), false, (B), (short)0, (C), false, false)

__device__ __forceinline__ int bin_of(float d) {
    int b = (int)floorf(d) + 64;
    return b < 0 ? 0 : (b > NBINS - 1 ? NBINS - 1 : b);
}

// per-wave tile compute: 2 q-tiles (wave, wave+8), bf16x3, bit-identical in both passes
__device__ __forceinline__ void gemm_tile(const __bf16* Ah, const __bf16* Al,
                                          const __bf16* Bh, const __bf16* Bl,
                                          int wave, int lane, v8f* o0, v8f* o1) {
    const int mrow = lane & 15;
    const int lh8  = (lane >> 4) * 8;
    const int lh16 = (lane >> 4) * 16;
    const __bf16* a0h = Ah + (size_t)((wave << 4) + mrow) * AST;
    const __bf16* a0l = Al + (size_t)((wave << 4) + mrow) * AST;
    const __bf16* a1h = Ah + (size_t)(((wave + 8) << 4) + mrow) * AST;
    const __bf16* a1l = Al + (size_t)(((wave + 8) << 4) + mrow) * AST;
    const __bf16* bh  = Bh + (size_t)mrow * AST;
    const __bf16* bl  = Bl + (size_t)mrow * AST;
    v8f c0 = {}, c1 = {};
    #pragma unroll
    for (int kc = 0; kc < DIMS; kc += 32) {
        v16bf vbh = ldB(bh, kc, lh16);
        v16bf vbl = ldB(bl, kc, lh16);
        {
            v16bf vah = ldA(a0h, kc, lh8);
            v16bf val = ldA(a0l, kc, lh8);
            c0 = WMMA_BF16(vah, vbh, c0);
            c0 = WMMA_BF16(vah, vbl, c0);
            c0 = WMMA_BF16(val, vbh, c0);
        }
        {
            v16bf vah = ldA(a1h, kc, lh8);
            v16bf val = ldA(a1l, kc, lh8);
            c1 = WMMA_BF16(vah, vbh, c1);
            c1 = WMMA_BF16(vah, vbl, c1);
            c1 = WMMA_BF16(val, vbh, c1);
        }
    }
    *o0 = c0;
    *o1 = c1;
}

// A staging: one float4 -> packed 4x bf16 hi + lo (8B stores)
__device__ __forceinline__ void stage_A(const float* qm, __bf16* As_hi, __bf16* As_lo, int t) {
    for (int f4 = t; f4 < NQ * (DIMS / 4); f4 += 256) {
        int qr = f4 >> 5;
        int c0 = (f4 & 31) * 4;
        float4 v = ((const float4*)qm)[qr * 32 + (f4 & 31)];
        __bf16 h[4], l[4];
        split2(v.x, &h[0], &l[0]);
        split2(v.y, &h[1], &l[1]);
        split2(v.z, &h[2], &l[2]);
        split2(v.w, &h[3], &l[3]);
        v2u ph = { pk(h[0], h[1]), pk(h[2], h[3]) };
        v2u pl = { pk(l[0], l[1]), pk(l[2], l[3]) };
        *(v2u*)&As_hi[qr * AST + c0] = ph;
        *(v2u*)&As_lo[qr * AST + c0] = pl;
    }
}

// B tile convert: 8 floats -> one b128 store each for hi and lo
__device__ __forceinline__ void convert_B(const float* raw, __bf16* bh, __bf16* bl, int t) {
    int r = t >> 4, c0 = (t & 15) * 8;
    const float4* s4 = (const float4*)&raw[r * DIMS + c0];
    float4 va = s4[0], vb = s4[1];
    __bf16 h[8], l[8];
    split2(va.x, &h[0], &l[0]);
    split2(va.y, &h[1], &l[1]);
    split2(va.z, &h[2], &l[2]);
    split2(va.w, &h[3], &l[3]);
    split2(vb.x, &h[4], &l[4]);
    split2(vb.y, &h[5], &l[5]);
    split2(vb.z, &h[6], &l[6]);
    split2(vb.w, &h[7], &l[7]);
    v4u ph = { pk(h[0], h[1]), pk(h[2], h[3]), pk(h[4], h[5]), pk(h[6], h[7]) };
    v4u pl = { pk(l[0], l[1]), pk(l[2], l[3]), pk(l[4], l[5]), pk(l[6], l[7]) };
    *(v4u*)&bh[r * AST + c0] = ph;
    *(v4u*)&bl[r * AST + c0] = pl;
}

// ---------------- init ----------------
__global__ void k_init(unsigned int* __restrict__ ghist,
                       unsigned int* __restrict__ ccnt,
                       float* __restrict__ preds) {
    int i = blockIdx.x * blockDim.x + threadIdx.x;
    int stride = gridDim.x * blockDim.x;
    for (int j = i; j < NQ * NBINS; j += stride) ghist[j] = 0u;
    for (int j = i; j < NQ; j += stride) ccnt[j] = 0u;
    for (int j = i; j < NQ * NCLS; j += stride) preds[j] = 0.0f;
}

// ---------------- pass 1: bf16x3 WMMA GEMM + per-query histogram ----------------
__global__ __launch_bounds__(256) void k_gemm_hist(const float* __restrict__ qm,
                                                   const float* __restrict__ mem,
                                                   unsigned int* __restrict__ ghist) {
    __shared__ __align__(16) __bf16 As_hi[NQ * AST];
    __shared__ __align__(16) __bf16 As_lo[NQ * AST];
    __shared__ __align__(16) float  Braw[2][16 * DIMS];
    __shared__ __align__(16) __bf16 Bs_hi[2][16 * AST];
    __shared__ __align__(16) __bf16 Bs_lo[2][16 * AST];
    __shared__ unsigned int hist[NQ * NBINS];

    const int t    = threadIdx.x;
    const int lane = t & 31;
    const int wave = t >> 5;

    for (int j = t; j < NQ * NBINS; j += 256) hist[j] = 0u;
    stage_A(qm, As_hi, As_lo, t);

    int nb = 0;
    int tile = blockIdx.x;
    if (tile < MT) {  // prefetch first tile (2x16B per thread = 512 chunks)
        const float* g = mem + (size_t)tile * 16 * DIMS;
        copy16_async(g + t * 4,         &Braw[0][t * 4]);
        copy16_async(g + (t + 256) * 4, &Braw[0][(t + 256) * 4]);
    }
    for (; tile < MT; tile += gridDim.x) {
        const int next = tile + gridDim.x;
        if (next < MT) {
            const float* g = mem + (size_t)next * 16 * DIMS;
            copy16_async(g + t * 4,         &Braw[nb ^ 1][t * 4]);
            copy16_async(g + (t + 256) * 4, &Braw[nb ^ 1][(t + 256) * 4]);
            async_wait_le2();               // current tile's copies done
        } else {
            async_wait_le0();
        }
        __syncthreads();
        convert_B(Braw[nb], Bs_hi[nb], Bs_lo[nb], t);
        __syncthreads();

        v8f acc0, acc1;
        gemm_tile(As_hi, As_lo, Bs_hi[nb], Bs_lo[nb], wave, lane, &acc0, &acc1);

        const int qoff0 = (wave << 4) + ((lane >> 4) << 3);
        const int qoff1 = ((wave + 8) << 4) + ((lane >> 4) << 3);
        #pragma unroll
        for (int r = 0; r < 8; ++r)
            atomicAdd(&hist[(qoff0 + r) * NBINS + bin_of(acc0[r])], 1u);
        #pragma unroll
        for (int r = 0; r < 8; ++r)
            atomicAdd(&hist[(qoff1 + r) * NBINS + bin_of(acc1[r])], 1u);
        nb ^= 1;
    }

    __syncthreads();
    for (int j = t; j < NQ * NBINS; j += 256) {
        unsigned int h = hist[j];
        if (h) atomicAdd(&ghist[j], h);
    }
}

// ---------------- pass 2: threshold bin per query ----------------
__global__ void k_scan(const unsigned int* __restrict__ ghist,
                       int* __restrict__ thresh, int* __restrict__ need) {
    int q = blockIdx.x * blockDim.x + threadIdx.x;
    if (q >= NQ) return;
    int s = 0, b = NBINS - 1;
    for (; b > 0; --b) {
        int h = (int)ghist[q * NBINS + b];
        if (s + h >= KNN) break;
        s += h;
    }
    thresh[q] = b;
    need[q]   = KNN - s;
}

// ---------------- pass 3: recompute; vote above bin, collect boundary bin ----------------
__global__ __launch_bounds__(256) void k_gemm_vote(const float* __restrict__ qm,
                                                   const float* __restrict__ mem,
                                                   const int* __restrict__ labels,
                                                   const int* __restrict__ thresh,
                                                   unsigned int* __restrict__ ccnt,
                                                   float* __restrict__ cval,
                                                   int* __restrict__ cidx,
                                                   float* __restrict__ preds) {
    __shared__ __align__(16) __bf16 As_hi[NQ * AST];
    __shared__ __align__(16) __bf16 As_lo[NQ * AST];
    __shared__ __align__(16) float  Braw[2][16 * DIMS];
    __shared__ __align__(16) __bf16 Bs_hi[2][16 * AST];
    __shared__ __align__(16) __bf16 Bs_lo[2][16 * AST];
    __shared__ int Tq[NQ];

    const int t    = threadIdx.x;
    const int lane = t & 31;
    const int wave = t >> 5;
    const int mrow = lane & 15;

    for (int j = t; j < NQ; j += 256) Tq[j] = thresh[j];
    stage_A(qm, As_hi, As_lo, t);

    int nb = 0;
    int tile = blockIdx.x;
    if (tile < MT) {
        const float* g = mem + (size_t)tile * 16 * DIMS;
        copy16_async(g + t * 4,         &Braw[0][t * 4]);
        copy16_async(g + (t + 256) * 4, &Braw[0][(t + 256) * 4]);
    }
    for (; tile < MT; tile += gridDim.x) {
        const int mbase = tile * 16;
        const int next  = tile + gridDim.x;
        if (next < MT) {
            const float* g = mem + (size_t)next * 16 * DIMS;
            copy16_async(g + t * 4,         &Braw[nb ^ 1][t * 4]);
            copy16_async(g + (t + 256) * 4, &Braw[nb ^ 1][(t + 256) * 4]);
            async_wait_le2();
        } else {
            async_wait_le0();
        }
        __syncthreads();
        convert_B(Braw[nb], Bs_hi[nb], Bs_lo[nb], t);
        __syncthreads();

        v8f acc0, acc1;
        gemm_tile(As_hi, As_lo, Bs_hi[nb], Bs_lo[nb], wave, lane, &acc0, &acc1);

        const int m = mbase + mrow;          // one memory row per lane
        int lbl = -1;                        // lazy label load
        #pragma unroll
        for (int half = 0; half < 2; ++half) {
            const v8f& acc = half ? acc1 : acc0;
            const int qoff = ((wave + half * 8) << 4) + ((lane >> 4) << 3);
            #pragma unroll
            for (int r = 0; r < 8; ++r) {
                float d  = acc[r];
                int   qi = qoff + r;
                int   b  = bin_of(d);
                int   T  = Tq[qi];
                if (b > T) {
                    if (lbl < 0) lbl = labels[m];
                    atomicAdd(&preds[qi * NCLS + lbl], __expf(d * INV_T));
                } else if (b == T) {
                    unsigned int pos = atomicAdd(&ccnt[qi], 1u);
                    if (pos < CAP) {
                        cval[qi * CAP + pos] = d;
                        cidx[qi * CAP + pos] = m;
                    }
                }
            }
        }
        nb ^= 1;
    }
}

// ---------------- pass 4: exact deterministic selection inside threshold bin ----------------
__global__ void k_final(const int* __restrict__ labels,
                        const int* __restrict__ need,
                        const unsigned int* __restrict__ ccnt,
                        const float* __restrict__ cval,
                        const int* __restrict__ cidx,
                        float* __restrict__ preds) {
    const int q = blockIdx.x;
    int c = (int)ccnt[q];
    if (c > CAP) c = CAP;
    const int nd = need[q];
    for (int i = threadIdx.x; i < c; i += blockDim.x) {
        float di = cval[q * CAP + i];
        int   mi = cidx[q * CAP + i];
        int rank = 0;
        for (int j = 0; j < c; ++j) {
            float dj = cval[q * CAP + j];
            int   mj = cidx[q * CAP + j];
            if (dj > di || (dj == di && mj < mi)) ++rank;
        }
        if (rank < nd) {
            atomicAdd(&preds[q * NCLS + labels[mi]], __expf(di * INV_T));
        }
    }
}

// ---------------- host launcher ----------------
extern "C" void kernel_launch(void* const* d_in, const int* in_sizes, int n_in,
                              void* d_out, int out_size, void* d_ws, size_t ws_size,
                              hipStream_t stream) {
    const float* qm     = (const float*)d_in[0];
    const float* mem    = (const float*)d_in[1];
    const int*   labels = (const int*)d_in[2];
    float* preds        = (float*)d_out;

    unsigned int* ghist  = (unsigned int*)d_ws;
    int*          thresh = (int*)(ghist + NQ * NBINS);
    int*          need   = thresh + NQ;
    unsigned int* ccnt   = (unsigned int*)(need + NQ);
    float*        cval   = (float*)(ccnt + NQ);
    int*          cidx   = (int*)(cval + NQ * CAP);

    k_init<<<64, 256, 0, stream>>>(ghist, ccnt, preds);
    k_gemm_hist<<<512, 256, 0, stream>>>(qm, mem, ghist);
    k_scan<<<1, 256, 0, stream>>>(ghist, thresh, need);
    k_gemm_vote<<<512, 256, 0, stream>>>(qm, mem, labels, thresh, ccnt, cval, cidx, preds);
    k_final<<<NQ, 128, 0, stream>>>(labels, need, ccnt, cval, cidx, preds);
}
